// MultiHeadAttentionLayer_70111046140070
// MI455X (gfx1250) — matrix operand order repaired and checked by reading, not compiled, not run on hardware
//
#include <hip/hip_runtime.h>
#include <hip/hip_bf16.h>

// MHA forward for gfx1250 (MI455X, wave32, WMMA, 320KB-LDS WGP).
// d_in order: query, key, value, mask, Wq, bq, Wk, bk, Wv, bv, Wo, bo
// d_out: [x (B*S*D) | attention (B*H*S*S)] f32
//
// Pipeline (HBM-traffic optimized: attention tensor touched exactly once):
//   1) f32->f16 convert of activations + weights (one pass)
//   2) Q/K/V projections, f16 outputs; V stored transposed [B,H,Dh,S]
//   3) fused scores -> masked softmax (in 128KB LDS stripe) -> attention store
//      -> attn@V, per 16-row q-stripe
//   4) output projection (f16 A via head-gather, f32 out + bias)

typedef __attribute__((ext_vector_type(16))) _Float16 v16h;
typedef __attribute__((ext_vector_type(8)))  _Float16 v8h;
typedef __attribute__((ext_vector_type(8)))  float    v8f;

constexpr int Bc = 4;
constexpr int Sc = 2048;
constexpr int Dc = 1024;
constexpr int Hc = 16;
constexpr int Dh = 64;
constexpr int Mc = Bc * Sc;            // 8192

// ---------------------------------------------------------------------------
// WMMA wrapper
// ---------------------------------------------------------------------------
__device__ __forceinline__ v8f wmma_f16(const v16h a, const v16h b, const v8f c) {
  return __builtin_amdgcn_wmma_f32_16x16x32_f16(
      false, a, false, b, (short)0, c, false, false);
}

// A fragment 16x32 (MxK) from row-major f16, leading dim lda.
// Lane L: m=L&15, hi=L>>4; VGPR v: K=(v<4?2v:16+2(v-4))+8*hi (+0,+1).
// v=0..3 -> 8 contiguous halves, v=4..7 -> 8 contiguous halves (b128-friendly).
__device__ __forceinline__ void load_a_h(v16h& a, const _Float16* __restrict__ A,
                                         int lda, int m_base, int k_base, int lane) {
  const int m  = m_base + (lane & 15);
  const int hi = lane >> 4;
  const _Float16* row = A + (size_t)m * lda + k_base + 8 * hi;
#pragma unroll
  for (int v = 0; v < 8; ++v) {
    const int k = (v < 4) ? (2 * v) : (16 + 2 * (v - 4));
    a[2 * v]     = row[k];
    a[2 * v + 1] = row[k + 1];
  }
}

// B fragment 32x16 (KxN) where memory holds rows of N with K contiguous
// (i.e. B = W^T for W[N][K], or V^T stored [Dh][S]).
// Lane L: n=L&15, hi=L>>4; halves K=8*hi+[0..7] and 16+8*hi+[0..7] (2x b128).
__device__ __forceinline__ void load_b_nk_h(v16h& b, const _Float16* __restrict__ W,
                                            int ldw, int n_base, int k_base, int lane) {
  const int n  = n_base + (lane & 15);
  const int hi = lane >> 4;
  const _Float16* row = W + (size_t)n * ldw + k_base + 8 * hi;
#pragma unroll
  for (int i = 0; i < 16; ++i) b[i] = row[i < 8 ? i : i + 8];
}

// A fragment from f32 LDS (softmax probabilities), ld = Sc, m_base = 0.
__device__ __forceinline__ void load_a_lds(v16h& a, const float* __restrict__ lds,
                                           int k_base, int lane) {
  const int m  = lane & 15;
  const int hi = lane >> 4;
  const float* row = lds + (size_t)m * Sc + k_base + 8 * hi;
#pragma unroll
  for (int v = 0; v < 8; ++v) {
    const int k = (v < 4) ? (2 * v) : (16 + 2 * (v - 4));
    a[2 * v]     = (_Float16)row[k];
    a[2 * v + 1] = (_Float16)row[k + 1];
  }
}

// C/D f32 store, row-major: lane L -> n=n_base+(L&15); VGPR r -> m=m_base+r+8*(L>>4)
__device__ __forceinline__ void store_c_rm_f32(float* __restrict__ C, int ldc,
                                               int m_base, int n_base, int lane,
                                               const v8f c, const float* __restrict__ bias) {
  const int n  = n_base + (lane & 15);
  const int hi = lane >> 4;
  const float bv = bias ? bias[n] : 0.0f;
#pragma unroll
  for (int r = 0; r < 8; ++r) {
    const int m = m_base + r + 8 * hi;
    C[(size_t)m * ldc + n] = c[r] + bv;
  }
}

// C/D store to head-split f16 [B,H,S,Dh]; row m=b*S+s, col n=h*Dh+dh
__device__ __forceinline__ void store_c_headsplit_h(_Float16* __restrict__ Y,
                                                    const float* __restrict__ bias,
                                                    int m_base, int n_base, int lane,
                                                    const v8f c) {
  const int n  = n_base + (lane & 15);
  const int hi = lane >> 4;
  const int h  = n >> 6;
  const int dh = n & 63;
  const float bv = bias[n];
#pragma unroll
  for (int r = 0; r < 8; ++r) {
    const int m = m_base + r + 8 * hi;
    const int b = m >> 11;
    const int s = m & 2047;
    Y[((((size_t)b * Hc + h) * Sc) + s) * Dh + dh] = (_Float16)(c[r] + bv);
  }
}

// C/D store to transposed head-split f16 [B,H,Dh,S] (for V)
__device__ __forceinline__ void store_c_vt_h(_Float16* __restrict__ Vt,
                                             const float* __restrict__ bias,
                                             int m_base, int n_base, int lane,
                                             const v8f c) {
  const int n  = n_base + (lane & 15);
  const int hi = lane >> 4;
  const int h  = n >> 6;
  const int dh = n & 63;
  const float bv = bias[n];
#pragma unroll
  for (int r = 0; r < 8; ++r) {
    const int m = m_base + r + 8 * hi;
    const int b = m >> 11;
    const int s = m & 2047;
    Vt[((((size_t)b * Hc + h) * Dh) + dh) * Sc + s] = (_Float16)(c[r] + bv);
  }
}

// ---------------------------------------------------------------------------
// Kernel 0: elementwise f32 -> f16 (8 elems/thread, b128 in / b128 out)
// ---------------------------------------------------------------------------
__global__ __launch_bounds__(256) void cvt_f32_f16_kernel(const float* __restrict__ in,
                                                          _Float16* __restrict__ out) {
  const size_t i = ((size_t)blockIdx.x * 256 + threadIdx.x) * 8;
  v8h o;
#pragma unroll
  for (int j = 0; j < 8; ++j) o[j] = (_Float16)in[i + j];
  *(v8h*)(out + i) = o;
}

// ---------------------------------------------------------------------------
// Kernel 1: projection (X16 @ W16^T + bias) -> head-split f16 (or transposed)
// One wave per block, 32x32 macro-tile (2x2 WMMA), K=1024.
// ---------------------------------------------------------------------------
template <bool TRANSPOSED>
__global__ __launch_bounds__(32) void mha_proj_kernel(const _Float16* __restrict__ X,
                                                      const _Float16* __restrict__ W,
                                                      const float* __restrict__ bias,
                                                      _Float16* __restrict__ Y) {
  const int lane = threadIdx.x;
  const int m0 = blockIdx.x * 32;
  const int n0 = blockIdx.y * 32;
  v8f c00 = {}, c01 = {}, c10 = {}, c11 = {};
  for (int k = 0; k < Dc; k += 32) {
    v16h a0, a1, b0, b1;
    load_a_h(a0, X, Dc, m0,      k, lane);
    load_a_h(a1, X, Dc, m0 + 16, k, lane);
    load_b_nk_h(b0, W, Dc, n0,      k, lane);
    load_b_nk_h(b1, W, Dc, n0 + 16, k, lane);
    c00 = wmma_f16(a0, b0, c00);
    c01 = wmma_f16(a0, b1, c01);
    c10 = wmma_f16(a1, b0, c10);
    c11 = wmma_f16(a1, b1, c11);
  }
  if (TRANSPOSED) {
    store_c_vt_h(Y, bias, m0,      n0,      lane, c00);
    store_c_vt_h(Y, bias, m0,      n0 + 16, lane, c01);
    store_c_vt_h(Y, bias, m0 + 16, n0,      lane, c10);
    store_c_vt_h(Y, bias, m0 + 16, n0 + 16, lane, c11);
  } else {
    store_c_headsplit_h(Y, bias, m0,      n0,      lane, c00);
    store_c_headsplit_h(Y, bias, m0,      n0 + 16, lane, c01);
    store_c_headsplit_h(Y, bias, m0 + 16, n0,      lane, c10);
    store_c_headsplit_h(Y, bias, m0 + 16, n0 + 16, lane, c11);
  }
}

// ---------------------------------------------------------------------------
// Kernel 2: fused scores -> masked softmax -> attention store -> attn@V.
// Block = 256 threads (8 waves), one (bh, 16-row q-stripe).
// Dynamic LDS: 16 x 2048 f32 = 128 KB score/probability stripe.
//   Stage A: each wave computes 8 16x32 score tiles (4 WMMA each) into LDS.
//   Stage B: each wave softmaxes 2 rows in LDS (float4 streaming), writes the
//            normalized row once to the attention output region of d_out.
//   Stage C: attn@V, N=64 as 4 16-wide tiles x 2-way K-split over 8 waves,
//            partials reduced through LDS, f16 result to XH workspace.
// ---------------------------------------------------------------------------
__global__ __launch_bounds__(256) void mha_fused_attn_kernel(
    const _Float16* __restrict__ Q16, const _Float16* __restrict__ K16,
    const _Float16* __restrict__ Vt16, const unsigned char* __restrict__ mask,
    float* __restrict__ attn_out, _Float16* __restrict__ XH16) {
  extern __shared__ float lds[];                 // 16*2048 floats
  const int tid  = threadIdx.x;
  const int wave = tid >> 5;
  const int lane = tid & 31;
  const int q0 = blockIdx.x * 16;
  const int bh = blockIdx.y;
  const int b  = bh >> 4;

  const _Float16* Qh = Q16  + (size_t)bh * Sc * Dh;
  const _Float16* Kh = K16  + (size_t)bh * Sc * Dh;
  const _Float16* Vh = Vt16 + (size_t)bh * Dh * Sc;   // [Dh][S]
  float* Aout = attn_out + ((size_t)bh * Sc + q0) * Sc;

  // ---- Stage A: scores (Q.K^T / 8) into LDS --------------------------------
  v16h aq0, aq1;
  load_a_h(aq0, Qh, Dh, q0, 0,  lane);
  load_a_h(aq1, Qh, Dh, q0, 32, lane);
#pragma unroll 1
  for (int pos = 0; pos < 8; ++pos) {
    const int n0 = (wave * 8 + pos) * 32;
    v8f c0 = {}, c1 = {};
    v16h b0, b1;
    load_b_nk_h(b0, Kh, Dh, n0,      0, lane);
    load_b_nk_h(b1, Kh, Dh, n0 + 16, 0, lane);
    c0 = wmma_f16(aq0, b0, c0);
    c1 = wmma_f16(aq0, b1, c1);
    load_b_nk_h(b0, Kh, Dh, n0,      32, lane);
    load_b_nk_h(b1, Kh, Dh, n0 + 16, 32, lane);
    c0 = wmma_f16(aq1, b0, c0);
    c1 = wmma_f16(aq1, b1, c1);
    const int n  = n0 + (lane & 15);
    const int hi = lane >> 4;
#pragma unroll
    for (int r = 0; r < 8; ++r) {
      const int m = r + 8 * hi;
      lds[(size_t)m * Sc + n]      = c0[r] * 0.125f;
      lds[(size_t)m * Sc + n + 16] = c1[r] * 0.125f;
    }
  }
  __syncthreads();

  // ---- Stage B: masked softmax in LDS, single write of attention -----------
  const uchar4* m4 = (const uchar4*)(mask + (size_t)b * Sc);
#pragma unroll 1
  for (int rr = 0; rr < 2; ++rr) {
    const int row = wave * 2 + rr;
    float* prow = lds + (size_t)row * Sc;
    float lmax = -3.0e38f;
    for (int it = 0; it < 16; ++it) {
      const int i4 = it * 32 + lane;
      float4 e = ((const float4*)prow)[i4];
      const uchar4 mm = m4[i4];
      e.x = mm.x ? e.x : -10000.0f;
      e.y = mm.y ? e.y : -10000.0f;
      e.z = mm.z ? e.z : -10000.0f;
      e.w = mm.w ? e.w : -10000.0f;
      lmax = fmaxf(lmax, fmaxf(fmaxf(e.x, e.y), fmaxf(e.z, e.w)));
    }
#pragma unroll
    for (int off = 16; off > 0; off >>= 1)
      lmax = fmaxf(lmax, __shfl_down(lmax, off, 32));
    lmax = __shfl(lmax, 0, 32);

    float lsum = 0.0f;
    for (int it = 0; it < 16; ++it) {
      const int i4 = it * 32 + lane;
      float4 e = ((const float4*)prow)[i4];
      const uchar4 mm = m4[i4];
      e.x = __expf((mm.x ? e.x : -10000.0f) - lmax);
      e.y = __expf((mm.y ? e.y : -10000.0f) - lmax);
      e.z = __expf((mm.z ? e.z : -10000.0f) - lmax);
      e.w = __expf((mm.w ? e.w : -10000.0f) - lmax);
      ((float4*)prow)[i4] = e;
      lsum += e.x + e.y + e.z + e.w;
    }
#pragma unroll
    for (int off = 16; off > 0; off >>= 1)
      lsum += __shfl_down(lsum, off, 32);
    lsum = __shfl(lsum, 0, 32);
    const float inv = 1.0f / lsum;

    float* grow = Aout + (size_t)row * Sc;
    for (int it = 0; it < 16; ++it) {
      const int i4 = it * 32 + lane;
      float4 p = ((const float4*)prow)[i4];
      p.x *= inv; p.y *= inv; p.z *= inv; p.w *= inv;
      ((float4*)prow)[i4] = p;
      ((float4*)grow)[i4] = p;     // only HBM touch of the attention tensor
    }
  }
  __syncthreads();

  // ---- Stage C: attn @ V (M=16, N=64, K=2048), 2-way K-split per tile ------
  const int t  = wave >> 1;             // 0..3 -> dh tile base
  const int kh = wave & 1;              // K half
  const int n0 = t * 16;
  v8f acc = {};
  for (int kk = kh * 1024; kk < kh * 1024 + 1024; kk += 32) {
    v16h af, bf;
    load_a_lds(af, lds, kk, lane);
    load_b_nk_h(bf, Vh, Sc, n0, kk, lane);   // V^T rows contiguous in K(=s)
    acc = wmma_f16(af, bf, acc);
  }
  __syncthreads();                           // all probability reads complete
  {
    const int n  = lane & 15;
    const int hi = lane >> 4;
    float* part = lds + wave * 256;
#pragma unroll
    for (int r = 0; r < 8; ++r) part[(r + 8 * hi) * 16 + n] = acc[r];
  }
  __syncthreads();
  if (wave < 4) {
    const float* p0 = lds + (2 * wave) * 256;
    const float* p1 = lds + (2 * wave + 1) * 256;
    _Float16* xh = XH16 + ((size_t)bh * Sc + q0) * Dh + wave * 16;
#pragma unroll
    for (int j = 0; j < 8; ++j) {
      const int i  = j * 32 + lane;          // 0..255
      const int ml = i >> 4;
      const int nl = i & 15;
      xh[(size_t)ml * Dh + nl] = (_Float16)(p0[i] + p1[i]);
    }
  }
}

// A fragment with head-gather indexing from f16 XH [B,H,S,Dh]:
// logical A[m,k]: m=b*S+s, k=h*64+dh. k pairs are even so never straddle heads.
__device__ __forceinline__ void load_a_hg_h(v16h& a, const _Float16* __restrict__ XH,
                                            int m_base, int k_base, int lane) {
  const int m  = m_base + (lane & 15);
  const int hi = lane >> 4;
  const int b  = m >> 11;
  const int s  = m & 2047;
  const _Float16* base = XH + ((size_t)b * Hc) * Sc * Dh + (size_t)s * Dh;
#pragma unroll
  for (int v = 0; v < 8; ++v) {
    const int k  = ((v < 4) ? (2 * v) : (16 + 2 * (v - 4))) + 8 * hi + k_base;
    const int h  = k >> 6;
    const int dh = k & 63;
    const _Float16* p = base + (size_t)h * Sc * Dh + dh;
    a[2 * v]     = p[0];
    a[2 * v + 1] = p[1];
  }
}

// ---------------------------------------------------------------------------
// Kernel 3: out = merge_heads(XH16) @ Wo16^T + bo   (f32 out)
// ---------------------------------------------------------------------------
__global__ __launch_bounds__(32) void mha_outproj_kernel(const _Float16* __restrict__ XH,
                                                         const _Float16* __restrict__ Wo,
                                                         const float* __restrict__ bo,
                                                         float* __restrict__ Out) {
  const int lane = threadIdx.x;
  const int m0 = blockIdx.x * 32;
  const int n0 = blockIdx.y * 32;
  v8f c00 = {}, c01 = {}, c10 = {}, c11 = {};
  for (int k = 0; k < Dc; k += 32) {
    v16h a0, a1, b0, b1;
    load_a_hg_h(a0, XH, m0,      k, lane);
    load_a_hg_h(a1, XH, m0 + 16, k, lane);
    load_b_nk_h(b0, Wo, Dc, n0,      k, lane);
    load_b_nk_h(b1, Wo, Dc, n0 + 16, k, lane);
    c00 = wmma_f16(a0, b0, c00);
    c01 = wmma_f16(a0, b1, c01);
    c10 = wmma_f16(a1, b0, c10);
    c11 = wmma_f16(a1, b1, c11);
  }
  store_c_rm_f32(Out, Dc, m0,      n0,      lane, c00, bo);
  store_c_rm_f32(Out, Dc, m0,      n0 + 16, lane, c01, bo);
  store_c_rm_f32(Out, Dc, m0 + 16, n0,      lane, c10, bo);
  store_c_rm_f32(Out, Dc, m0 + 16, n0 + 16, lane, c11, bo);
}

// ---------------------------------------------------------------------------
extern "C" void kernel_launch(void* const* d_in, const int* in_sizes, int n_in,
                              void* d_out, int out_size, void* d_ws, size_t ws_size,
                              hipStream_t stream) {
  (void)in_sizes; (void)n_in; (void)out_size; (void)ws_size;

  const float* query = (const float*)d_in[0];
  const float* key   = (const float*)d_in[1];
  const float* value = (const float*)d_in[2];
  const unsigned char* mask = (const unsigned char*)d_in[3];
  const float* Wq = (const float*)d_in[4];
  const float* bq = (const float*)d_in[5];
  const float* Wk = (const float*)d_in[6];
  const float* bk = (const float*)d_in[7];
  const float* Wv = (const float*)d_in[8];
  const float* bv = (const float*)d_in[9];
  const float* Wo = (const float*)d_in[10];
  const float* bo = (const float*)d_in[11];

  float* out_x = (float*)d_out;                         // [B,S,D] f32
  float* attn  = out_x + (size_t)Bc * Sc * Dc;          // [B,H,S,S] f32

  const size_t actsz  = (size_t)Mc * Dc;                // 8,388,608
  const size_t wsz    = (size_t)Dc * Dc;                // 1,048,576
  const size_t headsz = (size_t)Bc * Hc * Sc * Dh;      // 8,388,608

  _Float16* Xq16 = (_Float16*)d_ws;
  _Float16* Xk16 = Xq16 + actsz;
  _Float16* Xv16 = Xk16 + actsz;
  _Float16* Wq16 = Xv16 + actsz;
  _Float16* Wk16 = Wq16 + wsz;
  _Float16* Wv16 = Wk16 + wsz;
  _Float16* Wo16 = Wv16 + wsz;
  _Float16* Q16  = Wo16 + wsz;
  _Float16* K16  = Q16 + headsz;
  _Float16* Vt16 = K16 + headsz;
  _Float16* XH16 = Vt16 + headsz;

  // 0) convert activations + weights to f16
  {
    dim3 ga((unsigned)(actsz / (256 * 8)));             // 4096
    dim3 gw((unsigned)(wsz / (256 * 8)));               // 512
    cvt_f32_f16_kernel<<<ga, 256, 0, stream>>>(query, Xq16);
    cvt_f32_f16_kernel<<<ga, 256, 0, stream>>>(key,   Xk16);
    cvt_f32_f16_kernel<<<ga, 256, 0, stream>>>(value, Xv16);
    cvt_f32_f16_kernel<<<gw, 256, 0, stream>>>(Wq, Wq16);
    cvt_f32_f16_kernel<<<gw, 256, 0, stream>>>(Wk, Wk16);
    cvt_f32_f16_kernel<<<gw, 256, 0, stream>>>(Wv, Wv16);
    cvt_f32_f16_kernel<<<gw, 256, 0, stream>>>(Wo, Wo16);
  }

  // 1) projections: Q,K head-split f16; V transposed [B,H,Dh,S] f16
  {
    dim3 grid(Mc / 32, Dc / 32);                        // (256, 32)
    mha_proj_kernel<false><<<grid, 32, 0, stream>>>(Xq16, Wq16, bq, Q16);
    mha_proj_kernel<false><<<grid, 32, 0, stream>>>(Xk16, Wk16, bk, K16);
    mha_proj_kernel<true ><<<grid, 32, 0, stream>>>(Xv16, Wv16, bv, Vt16);
  }

  // 2) fused scores -> softmax -> attention store -> attn@V
  {
    dim3 grid(Sc / 16, Bc * Hc);                        // (128, 64)
    const size_t lds_bytes = (size_t)16 * Sc * sizeof(float);   // 128 KB
    mha_fused_attn_kernel<<<grid, 256, lds_bytes, stream>>>(
        Q16, K16, Vt16, mask, attn, XH16);
  }

  // 3) output projection
  {
    dim3 grid(Mc / 32, Dc / 32);                        // (256, 32)
    mha_outproj_kernel<<<grid, 32, 0, stream>>>(XH16, Wo16, bo, out_x);
  }
}
